// MockSharedExpertMoE_77773267796173
// MI455X (gfx1250) — matrix operand order repaired and checked by reading, not compiled, not run on hardware
//
#include <hip/hip_runtime.h>

// out = relu(x @ W_up^T) @ W_down^T + x @ W_expert   (gating collapses to 1.0)
// fp32 emulated via bf16 hi/lo split: x*w ~= xh*wh + xh*wl + xl*wh (3 bf16 WMMAs, f32 acc)
// Tile staging via Tensor Data Mover (tensor_load_to_lds) with double-buffered LDS.

typedef __bf16 bf16;
typedef __attribute__((ext_vector_type(16))) __bf16 v16bf;
typedef __attribute__((ext_vector_type(8)))  __bf16 v8bf;
typedef __attribute__((ext_vector_type(8)))  float  v8f;
typedef __attribute__((ext_vector_type(4)))  unsigned int u32x4;
typedef __attribute__((ext_vector_type(4)))  int i32x4;
typedef __attribute__((ext_vector_type(8)))  int i32x8;

#define DIM    2048
#define TOKENS (4 * 4096)          // 16384
#define PITCH  40                  // 32 cols + 8 pad bf16 -> 80B row pitch (bank-friendly)
#define TILE_ELEMS (128 * PITCH)

#if __has_builtin(__builtin_amdgcn_tensor_load_to_lds)
#define USE_TDM 1
#endif

// ---------------------------------------------------------------------------
// Conversion kernels
// ---------------------------------------------------------------------------
__global__ __launch_bounds__(256)
void split_fp32_bf16(const float* __restrict__ src, bf16* __restrict__ hi,
                     bf16* __restrict__ lo, int n) {
  int i = blockIdx.x * blockDim.x + threadIdx.x;
  if (i < n) {
    float v = src[i];
    bf16 h = (bf16)v;
    hi[i] = h;
    lo[i] = (bf16)(v - (float)h);
  }
}

// W_expert is used as x @ W (not W^T): store transposed so GEMM reads row n = K-contiguous.
__global__ __launch_bounds__(256)
void transpose_split_fp32_bf16(const float* __restrict__ src,  // [DIM][DIM] row-major
                               bf16* __restrict__ hiT, bf16* __restrict__ loT) {
  int i = blockIdx.x * blockDim.x + threadIdx.x;  // coalesced read
  int k = i / DIM, n = i % DIM;
  float v = src[i];
  bf16 h = (bf16)v;
  size_t o = (size_t)n * DIM + k;
  hiT[o] = h;
  loT[o] = (bf16)(v - (float)h);
}

// ---------------------------------------------------------------------------
// WMMA helpers
// ---------------------------------------------------------------------------
__device__ inline v8f wmma_bf16(v16bf a, v16bf b, v8f c) {
  return __builtin_amdgcn_wmma_f32_16x16x32_bf16(false, a, false, b, (short)0, c,
                                                 false, false);
}

// A-matrix 16x32 bf16 lane layout: lane L -> M = L&15, kb = L>>4,
// elements 0..7 = K[kb*8 ..], elements 8..15 = K[16+kb*8 ..]
__device__ inline v16bf load_fragA(const bf16* lds, int mt, int lane) {
  int m  = mt + (lane & 15);
  int kb = lane >> 4;
  v8bf p0 = *reinterpret_cast<const v8bf*>(lds + m * PITCH + kb * 8);
  v8bf p1 = *reinterpret_cast<const v8bf*>(lds + m * PITCH + 16 + kb * 8);
  v16bf r;
#pragma unroll
  for (int e = 0; e < 8; ++e) { r[e] = p0[e]; r[e + 8] = p1[e]; }
  return r;
}

// B-matrix 32x16 bf16: lane L -> N = L&15, kb = L>>4,
// elements 0..7 = K[kb*8 ..], elements 8..15 = K[16+kb*8 ..]
__device__ inline v16bf load_fragB(const bf16* lds, int nt, int lane) {
  int n  = nt + (lane & 15);
  int kb = lane >> 4;
  v8bf p0 = *reinterpret_cast<const v8bf*>(lds + n * PITCH + kb * 8);
  v8bf p1 = *reinterpret_cast<const v8bf*>(lds + n * PITCH + 16 + kb * 8);
  v16bf r;
#pragma unroll
  for (int e = 0; e < 8; ++e) { r[e] = p0[e]; r[e + 8] = p1[e]; }
  return r;
}

// One k-step of the 3-term split-bf16 MMA (24 WMMAs per wave)
__device__ inline void mma_step(const bf16* tAh, const bf16* tAl,
                                const bf16* tBh, const bf16* tBl,
                                int wm, int wn, int lane, v8f acc[4][2]) {
  v16bf ah[4], al[4], bh[2], bl[2];
#pragma unroll
  for (int i = 0; i < 4; ++i) {
    ah[i] = load_fragA(tAh, wm + 16 * i, lane);
    al[i] = load_fragA(tAl, wm + 16 * i, lane);
  }
#pragma unroll
  for (int j = 0; j < 2; ++j) {
    bh[j] = load_fragB(tBh, wn + 16 * j, lane);
    bl[j] = load_fragB(tBl, wn + 16 * j, lane);
  }
#pragma unroll
  for (int i = 0; i < 4; ++i)
#pragma unroll
    for (int j = 0; j < 2; ++j) {
      acc[i][j] = wmma_bf16(ah[i], bh[j], acc[i][j]);
      acc[i][j] = wmma_bf16(ah[i], bl[j], acc[i][j]);
      acc[i][j] = wmma_bf16(al[i], bh[j], acc[i][j]);
    }
}

// ---------------------------------------------------------------------------
// Tensor Data Mover: DMA a 128x32 bf16 tile (row-major, row stride DIM) into
// LDS with HW padding: 64B data + 16B pad per row -> PITCH=40 bf16 rows.
// ---------------------------------------------------------------------------
#ifdef USE_TDM
__device__ inline void tdm_load_tile(unsigned lds_off, const bf16* g,
                                     int row0, int k0, int rows_total) {
  unsigned long long ga =
      (unsigned long long)(uintptr_t)(g + (size_t)row0 * DIM + k0);
  u32x4 g0;
  g0[0] = 1u;                                   // count=1, no gather
  g0[1] = lds_off;                              // lds_addr (bytes)
  g0[2] = (unsigned)ga;                         // global_addr[31:0]
  g0[3] = (unsigned)(ga >> 32) | (2u << 30);    // global_addr[56:32] | type=2
  i32x8 g1;
  g1[0] = (1 << 16)     // data_size = 2 bytes
        | (1 << 20)     // pad_enable
        | (3 << 22)     // pad_interval: 64B of data
        | (3 << 25);    // pad_amount: 4 dwords (16B)
  g1[1] = (int)((unsigned)DIM << 16);           // tensor_dim0[15:0] = 2048
  g1[2] = (int)((unsigned)rows_total << 16);    // td0 hi=0 | tensor_dim1[15:0]
  g1[3] = (int)(32u << 16);                     // td1 hi=0 | tile_dim0 = 32
  g1[4] = 128;                                  // tile_dim1 = 128, tile_dim2 = 0
  g1[5] = DIM;                                  // tensor_dim0_stride = 2048
  g1[6] = 0;
  g1[7] = 0;
  i32x4 z4 = {0, 0, 0, 0};
#if __clang_major__ >= 23
  i32x8 z8 = {0, 0, 0, 0, 0, 0, 0, 0};
  __builtin_amdgcn_tensor_load_to_lds(g0, g1, z4, z4, z8, 0);
#else
  __builtin_amdgcn_tensor_load_to_lds(g0, g1, z4, z4, 0);
#endif
}
#endif

// Fallback cooperative tile load (256 threads, 2 x b128 each)
__device__ inline void load_tile(bf16* lds, const bf16* __restrict__ g,
                                 int row0, int k0, int tid) {
#pragma unroll
  for (int s = 0; s < 2; ++s) {
    int seg = tid + s * 256;
    int r   = seg >> 2;
    int c   = (seg & 3) * 8;
    v8bf v = *reinterpret_cast<const v8bf*>(g + (size_t)(row0 + r) * DIM + k0 + c);
    *reinterpret_cast<v8bf*>(lds + r * PITCH + c) = v;
  }
}

// ---------------------------------------------------------------------------
// GEMM1: H = relu(X @ Wup^T), H re-split into bf16 hi/lo for GEMM2
// ---------------------------------------------------------------------------
__global__ __launch_bounds__(256)
void gemm_up_relu_split(const bf16* __restrict__ Xh, const bf16* __restrict__ Xl,
                        const bf16* __restrict__ Wh, const bf16* __restrict__ Wl,
                        bf16* __restrict__ Hh, bf16* __restrict__ Hl) {
  __shared__ bf16 smem[2][4][TILE_ELEMS];

  int tid  = threadIdx.x;
  int lane = tid & 31;
  int wid  = tid >> 5;
  int wm   = (wid & 1) * 64;
  int wn   = (wid >> 1) * 32;
  int bM   = blockIdx.y * 128;
  int bN   = blockIdx.x * 128;

  v8f acc[4][2];
#pragma unroll
  for (int i = 0; i < 4; ++i)
#pragma unroll
    for (int j = 0; j < 2; ++j) acc[i][j] = (v8f)(0.0f);

#ifdef USE_TDM
  int t = __builtin_amdgcn_readfirstlane(wid);   // tile owned by this wave (0..3)
  const bf16* gsrc = (t == 0) ? Xh : (t == 1) ? Xl : (t == 2) ? Wh : Wl;
  int r0 = (t < 2) ? bM : bN;
  int rt = (t < 2) ? TOKENS : DIM;

  if (wid < 4) {                                 // prologue fetch, buffer 0
    tdm_load_tile((unsigned)(uintptr_t)&smem[0][t][0], gsrc, r0, 0, rt);
    __builtin_amdgcn_s_wait_tensorcnt(0);
  }
  __syncthreads();

  int buf = 0;
  for (int k0 = 0; k0 < DIM; k0 += 32) {
    if (k0 + 32 < DIM && wid < 4)                // prefetch next k-step
      tdm_load_tile((unsigned)(uintptr_t)&smem[buf ^ 1][t][0], gsrc, r0, k0 + 32, rt);

    mma_step(smem[buf][0], smem[buf][1], smem[buf][2], smem[buf][3],
             wm, wn, lane, acc);

    if (wid < 4) __builtin_amdgcn_s_wait_tensorcnt(0);
    __syncthreads();
    buf ^= 1;
  }
#else
  for (int k0 = 0; k0 < DIM; k0 += 32) {
    load_tile(smem[0][0], Xh, bM, k0, tid);
    load_tile(smem[0][1], Xl, bM, k0, tid);
    load_tile(smem[0][2], Wh, bN, k0, tid);
    load_tile(smem[0][3], Wl, bN, k0, tid);
    __syncthreads();
    mma_step(smem[0][0], smem[0][1], smem[0][2], smem[0][3], wm, wn, lane, acc);
    __syncthreads();
  }
#endif

  // C/D layout: VGPR r -> M = r (lanes 0-15) / 8+r (lanes 16-31); N = lane&15
  int mofs = bM + wm + ((lane >> 4) << 3);
#pragma unroll
  for (int i = 0; i < 4; ++i)
#pragma unroll
    for (int j = 0; j < 2; ++j) {
      int n = bN + wn + 16 * j + (lane & 15);
#pragma unroll
      for (int r = 0; r < 8; ++r) {
        float v = fmaxf(acc[i][j][r], 0.0f);     // relu
        bf16 h = (bf16)v;
        size_t o = (size_t)(mofs + 16 * i + r) * DIM + n;
        Hh[o] = h;
        Hl[o] = (bf16)(v - (float)h);
      }
    }
}

// ---------------------------------------------------------------------------
// GEMM2 (fused K=4096): out = H @ Wdown^T + X @ W_expert (WexpT pre-transposed)
// ---------------------------------------------------------------------------
__global__ __launch_bounds__(256)
void gemm_down_plus_expert(const bf16* __restrict__ Hh, const bf16* __restrict__ Hl,
                           const bf16* __restrict__ Wdh, const bf16* __restrict__ Wdl,
                           const bf16* __restrict__ Xh, const bf16* __restrict__ Xl,
                           const bf16* __restrict__ Weh, const bf16* __restrict__ Wel,
                           float* __restrict__ out) {
  __shared__ bf16 smem[2][4][TILE_ELEMS];

  int tid  = threadIdx.x;
  int lane = tid & 31;
  int wid  = tid >> 5;
  int wm   = (wid & 1) * 64;
  int wn   = (wid >> 1) * 32;
  int bM   = blockIdx.y * 128;
  int bN   = blockIdx.x * 128;

  v8f acc[4][2];
#pragma unroll
  for (int i = 0; i < 4; ++i)
#pragma unroll
    for (int j = 0; j < 2; ++j) acc[i][j] = (v8f)(0.0f);

#ifdef USE_TDM
  int t = __builtin_amdgcn_readfirstlane(wid);
  int r0 = (t < 2) ? bM : bN;
  int rt = (t < 2) ? TOKENS : DIM;

  if (wid < 4) {
    const bf16* g = (t == 0) ? Hh : (t == 1) ? Hl : (t == 2) ? Wdh : Wdl;
    tdm_load_tile((unsigned)(uintptr_t)&smem[0][t][0], g, r0, 0, rt);
    __builtin_amdgcn_s_wait_tensorcnt(0);
  }
  __syncthreads();

  int buf = 0;
  for (int k0 = 0; k0 < 2 * DIM; k0 += 32) {
    int nk = k0 + 32;
    if (nk < 2 * DIM && wid < 4) {
      const bf16* g;
      int kk;
      if (nk < DIM) {
        g  = (t == 0) ? Hh : (t == 1) ? Hl : (t == 2) ? Wdh : Wdl;
        kk = nk;
      } else {
        g  = (t == 0) ? Xh : (t == 1) ? Xl : (t == 2) ? Weh : Wel;
        kk = nk - DIM;
      }
      tdm_load_tile((unsigned)(uintptr_t)&smem[buf ^ 1][t][0], g, r0, kk, rt);
    }

    mma_step(smem[buf][0], smem[buf][1], smem[buf][2], smem[buf][3],
             wm, wn, lane, acc);

    if (wid < 4) __builtin_amdgcn_s_wait_tensorcnt(0);
    __syncthreads();
    buf ^= 1;
  }
#else
  for (int k0 = 0; k0 < 2 * DIM; k0 += 32) {
    const bf16 *ah_g, *al_g, *bh_g, *bl_g;
    int kk;
    if (k0 < DIM) { ah_g = Hh; al_g = Hl; bh_g = Wdh; bl_g = Wdl; kk = k0; }
    else          { ah_g = Xh; al_g = Xl; bh_g = Weh; bl_g = Wel; kk = k0 - DIM; }
    load_tile(smem[0][0], ah_g, bM, kk, tid);
    load_tile(smem[0][1], al_g, bM, kk, tid);
    load_tile(smem[0][2], bh_g, bN, kk, tid);
    load_tile(smem[0][3], bl_g, bN, kk, tid);
    __syncthreads();
    mma_step(smem[0][0], smem[0][1], smem[0][2], smem[0][3], wm, wn, lane, acc);
    __syncthreads();
  }
#endif

  int mofs = bM + wm + ((lane >> 4) << 3);
#pragma unroll
  for (int i = 0; i < 4; ++i)
#pragma unroll
    for (int j = 0; j < 2; ++j) {
      int n = bN + wn + 16 * j + (lane & 15);
#pragma unroll
      for (int r = 0; r < 8; ++r)
        out[(size_t)(mofs + 16 * i + r) * DIM + n] = acc[i][j][r];
    }
}

// ---------------------------------------------------------------------------
extern "C" void kernel_launch(void* const* d_in, const int* in_sizes, int n_in,
                              void* d_out, int out_size, void* d_ws, size_t ws_size,
                              hipStream_t stream) {
  const float* x   = (const float*)d_in[0];
  const float* Wup = (const float*)d_in[1];
  const float* Wdn = (const float*)d_in[2];
  // d_in[3] = W_gate: unused (softmax-of-topk weights sum to exactly 1)
  const float* Wex = (const float*)d_in[4];

  const size_t SZ_X = (size_t)TOKENS * DIM;
  const size_t SZ_W = (size_t)DIM * DIM;

  bf16* p    = (bf16*)d_ws;                   // ~304 MB bf16 scratch
  bf16* Xh   = p;           p += SZ_X;
  bf16* Xl   = p;           p += SZ_X;
  bf16* Hh   = p;           p += SZ_X;
  bf16* Hl   = p;           p += SZ_X;
  bf16* Wuph = p;           p += SZ_W;
  bf16* Wupl = p;           p += SZ_W;
  bf16* Wdh  = p;           p += SZ_W;
  bf16* Wdl  = p;           p += SZ_W;
  bf16* Weh  = p;           p += SZ_W;
  bf16* Wel  = p;           p += SZ_W;

  split_fp32_bf16<<<(int)(SZ_X / 256), 256, 0, stream>>>(x, Xh, Xl, (int)SZ_X);
  split_fp32_bf16<<<(int)(SZ_W / 256), 256, 0, stream>>>(Wup, Wuph, Wupl, (int)SZ_W);
  split_fp32_bf16<<<(int)(SZ_W / 256), 256, 0, stream>>>(Wdn, Wdh, Wdl, (int)SZ_W);
  transpose_split_fp32_bf16<<<(int)(SZ_W / 256), 256, 0, stream>>>(Wex, Weh, Wel);

  dim3 grid(DIM / 128, TOKENS / 128);

  gemm_up_relu_split<<<grid, 256, 0, stream>>>(Xh, Xl, Wuph, Wupl, Hh, Hl);
  gemm_down_plus_expert<<<grid, 256, 0, stream>>>(Hh, Hl, Wdh, Wdl,
                                                  Xh, Xl, Weh, Wel,
                                                  (float*)d_out);
}
